// BinaryLinearLayer_21517786153628
// MI455X (gfx1250) — 8-bit matrix operand order repaired and checked by reading, compiled, not run on hardware
//
#include <hip/hip_runtime.h>

typedef int   v16i __attribute__((ext_vector_type(16)));
typedef float v8f  __attribute__((ext_vector_type(8)));
typedef char  c16  __attribute__((ext_vector_type(16)));

#define M_DIM 8192
#define K_DIM 4096
#define N_DIM 2048

#define BM 128
#define BN 128
#define BK 128
#define KSTEPS (K_DIM / BK)

__device__ __forceinline__ char sgn_fp8(float v) {
    return v > 0.f ? (char)0x38 : (v < 0.f ? (char)0xB8 : (char)0x00);
}

__global__ __launch_bounds__(256) void pack_x_kernel(const float4* __restrict__ X4,
                                                     c16* __restrict__ Xq16) {
    const size_t t = (size_t)blockIdx.x * blockDim.x + threadIdx.x;
    c16 v;
#pragma unroll
    for (int j = 0; j < 4; ++j) {
        float4 f = X4[t * 4 + j];
        v[j * 4 + 0] = sgn_fp8(f.x);
        v[j * 4 + 1] = sgn_fp8(f.y);
        v[j * 4 + 2] = sgn_fp8(f.z);
        v[j * 4 + 3] = sgn_fp8(f.w);
    }
    Xq16[t] = v;
}

__global__ __launch_bounds__(256) void pack_w_kernel(const float* __restrict__ W,
                                                     char* __restrict__ Wq) {
    const size_t t  = (size_t)blockIdx.x * blockDim.x + threadIdx.x;
    const int    k  = (int)(t >> 9);
    const int    n4 = (int)(t & 511) * 4;
    float4 w = *(const float4*)(W + (size_t)k * N_DIM + n4);
    Wq[(size_t)(n4 + 0) * K_DIM + k] = sgn_fp8(w.x);
    Wq[(size_t)(n4 + 1) * K_DIM + k] = sgn_fp8(w.y);
    Wq[(size_t)(n4 + 2) * K_DIM + k] = sgn_fp8(w.z);
    Wq[(size_t)(n4 + 3) * K_DIM + k] = sgn_fp8(w.w);
}

__global__ __launch_bounds__(256) void bitgemm_kernel(const char* __restrict__ Xq,
                                                      const char* __restrict__ Wq,
                                                      const float* __restrict__ alpha,
                                                      const float* __restrict__ betta,
                                                      const float* __restrict__ gamma,
                                                      float* __restrict__ out) {
    __shared__ char As[2][BM * BK];
    __shared__ char Bs[2][BN * BK];

    const int tid  = threadIdx.x;
    const int wave = tid >> 5;
    const int lane = tid & 31;
    const int half = lane >> 4;
    const int lm   = lane & 15;

    const int m0 = blockIdx.y * BM;
    const int n0 = blockIdx.x * BN;

    const int wm = wave >> 1;
    const int wn = wave & 1;

    v8f acc[2][4];
#pragma unroll
    for (int mt = 0; mt < 2; ++mt)
#pragma unroll
        for (int nt = 0; nt < 4; ++nt)
            acc[mt][nt] = (v8f){0.f, 0.f, 0.f, 0.f, 0.f, 0.f, 0.f, 0.f};

    const int lr = tid >> 3;
    const int lc = (tid & 7) * 16;

    int4 atmp[4], btmp[4];

#pragma unroll
    for (int p = 0; p < 4; ++p) {
        atmp[p] = *(const int4*)(Xq + (size_t)(m0 + lr + p * 32) * K_DIM + lc);
        btmp[p] = *(const int4*)(Wq + (size_t)(n0 + lr + p * 32) * K_DIM + lc);
    }
#pragma unroll
    for (int p = 0; p < 4; ++p) {
        *(int4*)(&As[0][(size_t)(lr + p * 32) * BK + lc]) = atmp[p];
        *(int4*)(&Bs[0][(size_t)(lr + p * 32) * BK + lc]) = btmp[p];
    }
    __syncthreads();

    for (int i = 0; i < KSTEPS; ++i) {
        const int cur = i & 1;
        const int nxt = cur ^ 1;
        const bool has_next = (i + 1) < KSTEPS;

        if (has_next) {
            const int k0 = (i + 1) * BK;
#pragma unroll
            for (int p = 0; p < 4; ++p) {
                atmp[p] = *(const int4*)(Xq + (size_t)(m0 + lr + p * 32) * K_DIM + k0 + lc);
                btmp[p] = *(const int4*)(Wq + (size_t)(n0 + lr + p * 32) * K_DIM + k0 + lc);
            }
        }

        v16i afrag[2];
#pragma unroll
        for (int mt = 0; mt < 2; ++mt) {
            const char* base = &As[cur][(size_t)(wm * 32 + mt * 16 + lm) * BK + half * 16];
#pragma unroll
            for (int j = 0; j < 8; ++j) {
                int2 q = *(const int2*)(base + (j >> 1) * 32 + (j & 1) * 8);
                afrag[mt][2 * j + 0] = q.x;
                afrag[mt][2 * j + 1] = q.y;
            }
        }

#pragma unroll
        for (int nt = 0; nt < 4; ++nt) {
            const char* base = &Bs[cur][(size_t)(wn * 64 + nt * 16 + lm) * BK + half * 16];
            v16i bfrag;
#pragma unroll
            for (int j = 0; j < 4; ++j) {
                int4 q = *(const int4*)(base + j * 32);
                bfrag[4 * j + 0] = q.x;
                bfrag[4 * j + 1] = q.y;
                bfrag[4 * j + 2] = q.z;
                bfrag[4 * j + 3] = q.w;
            }
#pragma unroll
            for (int mt = 0; mt < 2; ++mt) {
                acc[mt][nt] = __builtin_amdgcn_wmma_f32_16x16x128_fp8_fp8(
                    afrag[mt], bfrag, (short)0, acc[mt][nt], false, false);
            }
        }

        if (has_next) {
#pragma unroll
            for (int p = 0; p < 4; ++p) {
                *(int4*)(&As[nxt][(size_t)(lr + p * 32) * BK + lc]) = atmp[p];
                *(int4*)(&Bs[nxt][(size_t)(lr + p * 32) * BK + lc]) = btmp[p];
            }
        }
        __syncthreads();
    }

    const float a = fmaxf(alpha[0], 0.f);
#pragma unroll
    for (int nt = 0; nt < 4; ++nt) {
        const int ncol = n0 + wn * 64 + nt * 16 + lm;
        const float sc = a * fmaxf(betta[ncol >> 6], 0.f) * fmaxf(gamma[ncol & 63], 0.f);
#pragma unroll
        for (int mt = 0; mt < 2; ++mt) {
            const int mbase = m0 + wm * 32 + mt * 16 + half * 8;
#pragma unroll
            for (int r = 0; r < 8; ++r) {
                out[(size_t)(mbase + r) * N_DIM + ncol] = acc[mt][nt][r] * sc;
            }
        }
    }
}

extern "C" void kernel_launch(void* const* d_in, const int* in_sizes, int n_in,
                              void* d_out, int out_size, void* d_ws, size_t ws_size,
                              hipStream_t stream) {
    const float* X     = (const float*)d_in[0];
    const float* W     = (const float*)d_in[1];
    const float* alpha = (const float*)d_in[2];
    const float* betta = (const float*)d_in[3];
    const float* gamma = (const float*)d_in[4];
    float*       out   = (float*)d_out;

    char* Xq = (char*)d_ws;
    char* Wq = Xq + (size_t)M_DIM * K_DIM;

    pack_x_kernel<<<(M_DIM * (size_t)K_DIM) / 16 / 256, 256, 0, stream>>>(
        (const float4*)X, (c16*)Xq);
    pack_w_kernel<<<((size_t)K_DIM * (N_DIM / 4)) / 256, 256, 0, stream>>>(W, Wq);

    dim3 grid(N_DIM / BN, M_DIM / BM);
    bitgemm_kernel<<<grid, 256, 0, stream>>>(Xq, Wq, alpha, betta, gamma, out);
}
